// Support_AUG_31937376813210
// MI455X (gfx1250) — matrix operand order repaired and checked by reading, not compiled, not run on hardware
//
#include <hip/hip_runtime.h>
#include <math.h>

typedef __attribute__((ext_vector_type(16))) _Float16 v16h;
typedef __attribute__((ext_vector_type(8)))  float    v8f;
typedef __attribute__((ext_vector_type(4)))  int      v4i;

#define N_PATCH 39200      // 200*14*14
#define N_TILES 2450       // N_PATCH/16
#define HW      196        // 14*14
#define C_DIM   640
#define K_TILES 20         // C_DIM/32
#define M_SUP   980
#define M_PAD   992
#define M_TILES 62         // M_PAD/16
#define N_CLASS 5
#define SEL_NUM 784        // int(39200*0.02)
#define OUT_M   (M_SUP + SEL_NUM)   // 1764

#define TILE_U4 1280       // one A tile (20 kt * 32 lanes * 32B) = 20480B = 1280 x 16B

// -------- gfx1250 async global->LDS DMA (guarded; sync-copy fallback) -------
#if defined(__has_builtin)
#if __has_builtin(__builtin_amdgcn_global_load_async_to_lds_b128) && \
    __has_builtin(__builtin_amdgcn_s_wait_asynccnt)
#define USE_ASYNC_LDS 1
#endif
#endif

// pointer-to-AS1/AS3 vector types matching the builtin's parameter types
typedef __attribute__((address_space(1))) v4i* gv4i_p;
typedef __attribute__((address_space(3))) v4i* lv4i_p;

__device__ __forceinline__ void stage_tile(const uint4* __restrict__ gsrc,
                                           uint4* ldst, int tid) {
#ifdef USE_ASYNC_LDS
  // generic->AS1 is value-preserving; generic->AS3 keeps the low 32 bits,
  // which on AMDGPU is exactly the LDS offset.
  gv4i_p g = (gv4i_p)(unsigned long long)gsrc;
  lv4i_p l = (lv4i_p)(unsigned int)(unsigned long long)ldst;
#pragma unroll
  for (int i = 0; i < 5; ++i)
    __builtin_amdgcn_global_load_async_to_lds_b128(g + tid + i * 256,
                                                   l + tid + i * 256, 0, 0);
#else
#pragma unroll
  for (int i = 0; i < 5; ++i)
    ldst[tid + i * 256] = gsrc[tid + i * 256];
#endif
}

__device__ __forceinline__ void wait_stage() {
#ifdef USE_ASYNC_LDS
  __builtin_amdgcn_s_wait_asynccnt(0);
#endif
}

// ---------------------------------------------------------------------------
// Kernel 1: L2-normalize each patch column of raw=[C,N] and scatter into the
// CDNA5 WMMA B-matrix (32x16 f16) per-lane fragment layout.
// ufrag layout: [nt][kt][lane][16 halves]
// ---------------------------------------------------------------------------
__global__ __launch_bounds__(256) void build_unl_frags(
    const float* __restrict__ x1, _Float16* __restrict__ ufrag) {
  int n = blockIdx.x * blockDim.x + threadIdx.x;
  if (n >= N_PATCH) return;
  int b = n / HW, r = n % HW;
  const float* col = x1 + (size_t)b * C_DIM * HW + r;   // stride HW over c
  float ss = 0.f;
  for (int c = 0; c < C_DIM; ++c) { float v = col[(size_t)c * HW]; ss += v * v; }
  float inv = rsqrtf(ss);
  int nt = n >> 4, nl = n & 15;
  for (int c = 0; c < C_DIM; ++c) {
    float v  = col[(size_t)c * HW] * inv;
    int kt   = c >> 5, k = c & 31;
    int lane = nl + (((k >> 3) & 1) << 4);
    int idx  = ((k >> 4) << 3) + (k & 7);
    ufrag[(((size_t)(nt * K_TILES + kt)) * 32 + lane) * 16 + idx] = (_Float16)v;
  }
}

// ---------------------------------------------------------------------------
// Kernel 2: L2-normalize each support column of x2[j]=[C,M] and scatter into
// the CDNA5 WMMA A-matrix (16x32 f16) per-lane fragment layout.
// sfrag layout: [j][mt][kt][lane][16 halves]; columns m>=980 zero-padded.
// ---------------------------------------------------------------------------
__global__ __launch_bounds__(256) void build_sup_frags(
    const float* __restrict__ x2, _Float16* __restrict__ sfrag) {
  int t = blockIdx.x * blockDim.x + threadIdx.x;   // t = j*M_PAD + m
  if (t >= N_CLASS * M_PAD) return;
  int j = t / M_PAD, m = t % M_PAD;
  int mt = m >> 4, row = m & 15;
  const float* col = x2 + (size_t)j * C_DIM * M_SUP + m;
  float inv = 0.f;
  if (m < M_SUP) {
    float ss = 0.f;
    for (int c = 0; c < C_DIM; ++c) { float v = col[(size_t)c * M_SUP]; ss += v * v; }
    inv = rsqrtf(ss);
  }
  for (int c = 0; c < C_DIM; ++c) {
    float v = (m < M_SUP) ? col[(size_t)c * M_SUP] * inv : 0.f;
    int kt = c >> 5, k = c & 31;
    int G = k >> 4, kk = k & 15, h = kk >> 3, r2 = kk & 7;
    int p = (G << 2) + (r2 >> 1), w = r2 & 1;
    int lane = row + (h << 4);
    int idx  = (p << 1) + w;
    sfrag[(((size_t)((j * M_TILES + mt) * K_TILES + kt)) * 32 + lane) * 16 + idx] = (_Float16)v;
  }
}

// branchless running-top3 insert (maintains t0 >= t1 >= t2)
__device__ __forceinline__ void top3_insert(float v, float& t0, float& t1, float& t2) {
  float n0 = fmaxf(t0, v), m0 = fminf(t0, v);
  float n1 = fmaxf(t1, m0), m1 = fminf(t1, m0);
  float n2 = fmaxf(t2, m1);
  t0 = n0; t1 = n1; t2 = n2;
}

// ---------------------------------------------------------------------------
// Kernel 3 (the hot one): simT = supN^T @ unlN via v_wmma_f32_16x16x32_f16.
// 8 waves / WG, one 16-patch tile per wave; B fragments register-resident
// (20 x v16h). The A-tile sequence is identical for all 8 waves, so it is
// cooperatively staged global->LDS with async DMA, double-buffered behind
// the WMMA loop (s_wait_asynccnt + barrier). A fragments are software-
// pipelined one kt ahead so ds_load latency overlaps the WMMA pipe.
// Top-3 over M fused into the epilogue. Writes patch_to_class[j][n].
// ---------------------------------------------------------------------------
__global__ __launch_bounds__(256) void sim_top3_kernel(
    const _Float16* __restrict__ ufrag, const _Float16* __restrict__ sfrag,
    float* __restrict__ p2c) {
  __shared__ uint4 Abuf[2][TILE_U4];          // 2 x 20KB double buffer

  int tid  = threadIdx.x;
  int wave = tid >> 5;
  int lane = tid & 31;
  int nt_raw = blockIdx.x * 8 + wave;
  int nt = (nt_raw < N_TILES) ? nt_raw : (N_TILES - 1);  // keep all waves alive
  int j = blockIdx.y;

  // register-resident B fragments for this patch tile
  const v16h* ub = (const v16h*)ufrag + (size_t)nt * K_TILES * 32 + lane;
  v16h barr[K_TILES];
#pragma unroll
  for (int kt = 0; kt < K_TILES; ++kt) barr[kt] = ub[kt * 32];

  const uint4* agbase = (const uint4*)sfrag + (size_t)j * M_TILES * TILE_U4;

  float t0 = -INFINITY, t1 = -INFINITY, t2 = -INFINITY;
  int h8 = (lane >> 4) << 3;                  // C-layout: acc[v] holds row v + 8*(lane/16)

  stage_tile(agbase, &Abuf[0][0], tid);       // prologue: tile 0
  wait_stage();
  __syncthreads();

  for (int mt = 0; mt < M_TILES; ++mt) {
    int cur = mt & 1;
    if (mt + 1 < M_TILES)                     // async-prefetch next tile
      stage_tile(agbase + (size_t)(mt + 1) * TILE_U4, &Abuf[cur ^ 1][0], tid);

    const v16h* la = (const v16h*)&Abuf[cur][0] + lane;
    v8f acc = {0.f, 0.f, 0.f, 0.f, 0.f, 0.f, 0.f, 0.f};
    v16h a0 = la[0];                          // A-fragment pipeline, 1 deep
#pragma unroll
    for (int kt = 0; kt < K_TILES; ++kt) {
      v16h a1 = a0;
      if (kt + 1 < K_TILES) a1 = la[(kt + 1) * 32];
      acc = __builtin_amdgcn_wmma_f32_16x16x32_f16(
          false, a0, false, barr[kt], (short)0, acc, false, false);
      a0 = a1;
    }

    int mbase = mt * 16 + h8;
#pragma unroll
    for (int v = 0; v < 8; ++v) {
      float val = (mbase + v < M_SUP) ? acc[v] : -INFINITY;  // mask pad columns
      top3_insert(val, t0, t1, t2);
    }

    if (mt + 1 < M_TILES) {                   // staged buffer ready; all waves done reading cur
      wait_stage();
      __syncthreads();
    }
  }

  // merge the two lane-halves holding the same patch column
  float o0 = __shfl_xor(t0, 16, 32);
  float o1 = __shfl_xor(t1, 16, 32);
  float o2 = __shfl_xor(t2, 16, 32);
  top3_insert(o0, t0, t1, t2);
  top3_insert(o1, t0, t1, t2);
  top3_insert(o2, t0, t1, t2);

  int n = nt * 16 + (lane & 15);
  if (lane < 16 && nt_raw < N_TILES && n < N_PATCH)
    p2c[(size_t)j * N_PATCH + n] = t0 + t1 + t2;
}

// ---------------------------------------------------------------------------
// Kernel 4: exact top-784 selection with jax.lax.top_k ordering/tie-breaking.
// rank(e) = #{v_i > v_e} + #{i<e : v_i == v_e}; ranks are a permutation, so
// each rank<784 gets exactly one writer: sel[j][rank] = e.
// ---------------------------------------------------------------------------
#define SEL_TILE 2048
__global__ __launch_bounds__(256) void select_kernel(
    const float* __restrict__ p2c, int* __restrict__ sel) {
  __shared__ float tile[SEL_TILE];
  int j = blockIdx.y;
  int e = blockIdx.x * blockDim.x + threadIdx.x;
  const float* base = p2c + (size_t)j * N_PATCH;
  float ve = (e < N_PATCH) ? base[e] : -INFINITY;
  int rank = 0;
  for (int start = 0; start < N_PATCH; start += SEL_TILE) {
    int cnt = (N_PATCH - start < SEL_TILE) ? (N_PATCH - start) : SEL_TILE;
    __syncthreads();
    for (int i = threadIdx.x; i < cnt; i += blockDim.x) tile[i] = base[start + i];
    __syncthreads();
    for (int i = 0; i < cnt; ++i) {
      float vi = tile[i];
      rank += (vi > ve) || ((vi == ve) && ((start + i) < e));
    }
  }
  if (e < N_PATCH && rank < SEL_NUM) sel[(size_t)j * SEL_NUM + rank] = e;
}

// ---------------------------------------------------------------------------
// Kernel 5: out[j][c][:] = concat(x2[j][c][:], raw[c][sel[j][:]])
// ---------------------------------------------------------------------------
__global__ __launch_bounds__(256) void assemble_kernel(
    const float* __restrict__ x1, const float* __restrict__ x2,
    const int* __restrict__ sel, float* __restrict__ out) {
  size_t idx = (size_t)blockIdx.x * blockDim.x + threadIdx.x;
  const size_t total = (size_t)N_CLASS * C_DIM * OUT_M;
  if (idx >= total) return;
  int col = (int)(idx % OUT_M);
  size_t t = idx / OUT_M;
  int c = (int)(t % C_DIM);
  int j = (int)(t / C_DIM);
  float v;
  if (col < M_SUP) {
    v = x2[((size_t)j * C_DIM + c) * M_SUP + col];
  } else {
    int n = sel[(size_t)j * SEL_NUM + (col - M_SUP)];
    int b = n / HW, r = n % HW;
    v = x1[((size_t)b * C_DIM + c) * HW + r];
  }
  out[idx] = v;
}

extern "C" void kernel_launch(void* const* d_in, const int* in_sizes, int n_in,
                              void* d_out, int out_size, void* d_ws, size_t ws_size,
                              hipStream_t stream) {
  const float* x1 = (const float*)d_in[0];   // [200,640,14,14]
  const float* x2 = (const float*)d_in[1];   // [5,640,980]
  float* out = (float*)d_out;                // [5,640,1764]

  // workspace carving (all offsets 256B aligned)
  char* ws = (char*)d_ws;
  size_t off = 0;
  auto carve = [&](size_t bytes) { void* p = ws + off; off = (off + bytes + 255) & ~(size_t)255; return p; };
  _Float16* ufrag = (_Float16*)carve((size_t)N_TILES * K_TILES * 32 * 16 * sizeof(_Float16)); // ~50.2 MB
  _Float16* sfrag = (_Float16*)carve((size_t)N_CLASS * M_TILES * K_TILES * 32 * 16 * sizeof(_Float16)); // ~6.35 MB
  float*    p2c   = (float*)carve((size_t)N_CLASS * N_PATCH * sizeof(float));  // 784 KB
  int*      sel   = (int*)carve((size_t)N_CLASS * SEL_NUM * sizeof(int));      // 15.7 KB
  (void)ws_size;

  build_unl_frags<<<(N_PATCH + 255) / 256, 256, 0, stream>>>(x1, ufrag);
  build_sup_frags<<<(N_CLASS * M_PAD + 255) / 256, 256, 0, stream>>>(x2, sfrag);

  dim3 gGemm((N_TILES + 7) / 8, N_CLASS);
  sim_top3_kernel<<<gGemm, 256, 0, stream>>>(ufrag, sfrag, p2c);

  dim3 gSel((N_PATCH + 255) / 256, N_CLASS);
  select_kernel<<<gSel, 256, 0, stream>>>(p2c, sel);

  size_t total = (size_t)N_CLASS * C_DIM * OUT_M;
  assemble_kernel<<<(unsigned)((total + 255) / 256), 256, 0, stream>>>(x1, x2, sel, out);
  (void)in_sizes; (void)n_in; (void)out_size;
}
